// CustomMamba2D_28089086116527
// MI455X (gfx1250) — matrix operand order repaired and checked by reading, not compiled, not run on hardware
//
#include <hip/hip_runtime.h>

// ---------------------------------------------------------------------------
// CustomMamba2D fused pipeline for MI455X (gfx1250), wave32, WMMA f16->f32.
//   K1: xp = silu(bn(w_in @ x))        [TDM tensor_load_to_lds + WMMA]
//   K2: chunk partial sums of (wB @ xc) * exp(A*k)
//   K3: exclusive scan over chunk partials
//   K4: in-chunk scan -> state; wC@state + D*xc; gate; out-proj [WMMA heavy]
// ---------------------------------------------------------------------------

typedef __attribute__((ext_vector_type(16))) _Float16 v16h;
typedef __attribute__((ext_vector_type(8)))  float    v8f;
typedef __attribute__((ext_vector_type(4)))  unsigned int u32x4;
typedef __attribute__((ext_vector_type(8)))  int      i32x8;
typedef __attribute__((ext_vector_type(4)))  int      i32x4;

#define NB      8
#define DMODEL  64
#define DINNER  64
#define DSTATE  16
#define HH      512
#define WWID    512
#define LL      (HH*WWID)      // 262144
#define CHUNK   128
#define NCHROW  (WWID/CHUNK)   // 4
#define NCH     (LL/CHUNK)     // 2048 chunks per batch (contiguous in k)

__device__ __forceinline__ float sigm(float x) { return 1.f / (1.f + __expf(-x)); }
__device__ __forceinline__ float silu(float x) { return x * sigm(x); }

// A-matrix (16x32 f16) lane/element -> K mapping (ISA 7.12.2)
__device__ __forceinline__ int kmapA(int l, int e) {
    int base = (l & 16) ? 8 : 0;
    return (e < 8) ? (base + e) : (16 + base + (e - 8));
}
// B-matrix (32x16 f16) lane/element -> K mapping
__device__ __forceinline__ int kmapB(int l, int e) {
    int base = (l & 16) ? 8 : 0;
    return (e < 8) ? (base + e) : (16 + base + (e - 8));
}

// ===========================================================================
// K1: in_proj (64x64 GEMM over pixels) + BN + SiLU -> xp (f16)
// grid: NB*NCH blocks, 256 threads (8 waves), 128 px per block
// Staging of the 64ch x 128px f32 tile is done by the Tensor Data Mover:
// one 2D D# (tile_dim0=128 contiguous px, tile_dim1=64 ch, stride0=LL),
// issued by wave 0, completion tracked with TENSORcnt.
// ===========================================================================
__global__ void k1_inproj(const float* __restrict__ x,
                          const float* __restrict__ w_in,
                          const float* __restrict__ g_in,
                          const float* __restrict__ b_in,
                          _Float16* __restrict__ xp)
{
    __shared__ float    xs32[DMODEL][CHUNK];    // 32 KB (TDM destination)
    __shared__ _Float16 xs[DMODEL][CHUNK];      // 16 KB (f16 feed for WMMA)
    __shared__ _Float16 wh[DMODEL * DMODEL];    // 8 KB

    int tid = threadIdx.x;
    int bat = blockIdx.x >> 11;
    int p0  = (blockIdx.x & (NCH - 1)) * CHUNK;
    const float* xb = x + (size_t)bat * DMODEL * LL;

#if defined(__gfx1250__) && __has_builtin(__builtin_amdgcn_tensor_load_to_lds)
    if (tid < 32) {
        // ---- D# group 0 (128b): count=1, lds_addr, global_addr, type=2 ----
        unsigned int lds_addr = (unsigned int)(size_t)(&xs32[0][0]);
        unsigned long long ga = (unsigned long long)(size_t)(xb + p0);
        u32x4 g0;
        g0[0] = 1u;                                        // count=1, flags=0
        g0[1] = lds_addr;                                  // bits 63:32
        g0[2] = (unsigned int)(ga & 0xFFFFFFFFu);          // bits 95:64
        g0[3] = (unsigned int)((ga >> 32) & 0x1FFFFFFu)    // bits 120:96
              | (2u << 30);                                // type=2 (image)
        // ---- D# group 1 (256b): sizes/strides ----
        i32x8 g1;
        g1[0] = (2 << 16);          // workgroup_mask=0, data_size=2 (4B)
        g1[1] = (CHUNK << 16);      // tensor_dim0[15:0] (=128) in bits 63:48
        g1[2] = (DMODEL << 16);     // tensor_dim0 hi=0, tensor_dim1=64
        g1[3] = (CHUNK << 16);      // tensor_dim1 hi=0, tile_dim0=128
        g1[4] = DMODEL;             // tile_dim1=64, tile_dim2=0
        g1[5] = LL;                 // tensor_dim0_stride lo (=262144)
        g1[6] = 0;                  // stride0 hi=0, stride1 lo=0
        g1[7] = 0;                  // stride1 hi=0
        i32x4 gz = {0, 0, 0, 0};    // groups 2/3: NULL (2D tensor)
#if __clang_major__ >= 23
        i32x8 gz8 = {0, 0, 0, 0, 0, 0, 0, 0};
        __builtin_amdgcn_tensor_load_to_lds(g0, g1, gz, gz, gz8, 0);
#else
        __builtin_amdgcn_tensor_load_to_lds(g0, g1, gz, gz, 0);
#endif
    }
    // overlap: stage weights while the TDM runs
    for (int idx = tid; idx < DMODEL * DMODEL; idx += 256)
        wh[idx] = (_Float16)w_in[idx];
#if __has_builtin(__builtin_amdgcn_s_wait_tensorcnt)
    if (tid < 32) __builtin_amdgcn_s_wait_tensorcnt(0);
#endif
    __syncthreads();
#else
    // fallback: cooperative staging
    for (int idx = tid; idx < DMODEL * CHUNK; idx += 256) {
        int c = idx >> 7, px = idx & 127;
        xs32[c][px] = xb[(size_t)c * LL + p0 + px];
    }
    for (int idx = tid; idx < DMODEL * DMODEL; idx += 256)
        wh[idx] = (_Float16)w_in[idx];
    __syncthreads();
#endif

    // convert staged f32 tile -> f16 (keeps ds_load_b128-friendly frag feed)
    for (int idx = tid; idx < DMODEL * CHUNK; idx += 256) {
        int c = idx >> 7, px = idx & 127;
        xs[c][px] = (_Float16)xs32[c][px];
    }
    __syncthreads();

    int l = tid & 31, wid = tid >> 5;
    int n = l & 15, half = (l >> 4) & 1;
    int px0 = wid * 16;
    float bnr = rsqrtf(1.f + 1e-5f);

    v16h b0, b1;
#pragma unroll
    for (int e = 0; e < 16; ++e) {
        int kb = kmapB(l, e);
        b0[e] = xs[kb][px0 + n];
        b1[e] = xs[32 + kb][px0 + n];
    }
#pragma unroll
    for (int mt = 0; mt < 4; ++mt) {
        v16h a0, a1;
        int row = mt * 16 + n;
#pragma unroll
        for (int e = 0; e < 16; ++e) {
            int ka = kmapA(l, e);
            a0[e] = wh[row * 64 + ka];
            a1[e] = wh[row * 64 + 32 + ka];
        }
        v8f acc = {};
        acc = __builtin_amdgcn_wmma_f32_16x16x32_f16(false, a0, false, b0, (short)0, acc, false, false);
        acc = __builtin_amdgcn_wmma_f32_16x16x32_f16(false, a1, false, b1, (short)0, acc, false, false);
#pragma unroll
        for (int v = 0; v < 8; ++v) {
            int o = mt * 16 + half * 8 + v;
            float val = acc[v] * (g_in[o] * bnr) + b_in[o];
            val = silu(val);
            xp[(size_t)(bat * DINNER + o) * LL + p0 + px0 + n] = (_Float16)val;
        }
    }
}

// ===========================================================================
// K2: per-chunk partial sums of (wB @ xc) * exp(A*k)
// grid: NB*NCH, 256 threads
// ===========================================================================
__global__ void k2_partials(const _Float16* __restrict__ xp,
                            const float* __restrict__ w_dw,
                            const float* __restrict__ g_conv,
                            const float* __restrict__ b_conv,
                            const float* __restrict__ wB,
                            const float* __restrict__ A,
                            float* __restrict__ partials)
{
    __shared__ _Float16 xph[16][3][130];     // 12.2 KB halo stage
    __shared__ _Float16 xch[DINNER][CHUNK];  // 16 KB
    __shared__ float    red[DSTATE][CHUNK];  // 8 KB
    __shared__ float    wBs[DSTATE * 64];    // 4 KB
    __shared__ float    wdws[64 * 9];
    __shared__ float    scc[64], bcc[64], asd[16];

    int tid = threadIdx.x;
    int bat = blockIdx.x >> 11;
    int ch  = blockIdx.x & (NCH - 1);
    int h   = ch >> 2;
    int w0  = (ch & (NCHROW - 1)) * CHUNK;
    float bnr = rsqrtf(1.f + 1e-5f);

    for (int idx = tid; idx < DSTATE * 64; idx += 256) wBs[idx] = wB[idx];
    for (int idx = tid; idx < 64 * 9; idx += 256) wdws[idx] = w_dw[idx];
    if (tid < 64) { scc[tid] = g_conv[tid] * bnr; bcc[tid] = b_conv[tid]; }
    if (tid < 16) asd[tid] = A[tid];
    __syncthreads();

    for (int g = 0; g < 4; ++g) {
        for (int idx = tid; idx < 16 * 3 * 130; idx += 256) {
            int cg = idx / 390, r = idx % 390;
            int ry = r / 130, wx = r % 130;
            int c = g * 16 + cg;
            int hh = h + ry - 1, w = w0 + wx - 1;
            _Float16 v = (_Float16)0.f;
            if (hh >= 0 && hh < HH && w >= 0 && w < WWID)
                v = xp[(size_t)(bat * DINNER + c) * LL + (size_t)hh * WWID + w];
            xph[cg][ry][wx] = v;
        }
        __syncthreads();
        for (int idx = tid; idx < 16 * CHUNK; idx += 256) {
            int cg = idx >> 7, px = idx & 127;
            int c = g * 16 + cg;
            float s = 0.f;
#pragma unroll
            for (int dy = 0; dy < 3; ++dy)
#pragma unroll
                for (int dx = 0; dx < 3; ++dx)
                    s += (float)xph[cg][dy][px + dx] * wdws[c * 9 + dy * 3 + dx];
            s = silu(s * scc[c] + bcc[c]);
            xch[c][px] = (_Float16)s;
        }
        __syncthreads();
    }

    if (tid < CHUNK) {
        int px = tid;
        float kf = (float)(h * WWID + w0 + px);
#pragma unroll
        for (int s = 0; s < DSTATE; ++s) {
            float acc = 0.f;
            for (int c = 0; c < 64; ++c)
                acc += wBs[s * 64 + c] * (float)xch[c][px];
            red[s][px] = acc * __expf(asd[s] * kf);
        }
    }
    __syncthreads();
    for (int off = CHUNK / 2; off >= 1; off >>= 1) {
        for (int idx = tid; idx < DSTATE * off; idx += 256) {
            int s = idx / off, p = idx % off;
            red[s][p] += red[s][p + off];
        }
        __syncthreads();
    }
    if (tid < DSTATE)
        partials[(size_t)(bat * DSTATE + tid) * NCH + ch] = red[tid][0];
}

// ===========================================================================
// K3: exclusive scan over NCH chunk partials per (batch,state)
// grid: NB*DSTATE = 128 blocks, 256 threads, 8 elems/thread
// ===========================================================================
__global__ void k3_chunkscan(const float* __restrict__ partials,
                             float* __restrict__ offsets)
{
    __shared__ float sums[256];
    int tid = threadIdx.x;
    size_t base = (size_t)blockIdx.x * NCH;
    float vals[8];
    float lsum = 0.f;
#pragma unroll
    for (int j = 0; j < 8; ++j) {
        vals[j] = partials[base + tid * 8 + j];
        lsum += vals[j];
    }
    sums[tid] = lsum;
    __syncthreads();
    for (int off = 1; off < 256; off <<= 1) {
        float t = (tid >= off) ? sums[tid - off] : 0.f;
        __syncthreads();
        sums[tid] += t;
        __syncthreads();
    }
    float run = (tid == 0) ? 0.f : sums[tid - 1];
#pragma unroll
    for (int j = 0; j < 8; ++j) {
        offsets[base + tid * 8 + j] = run;
        run += vals[j];
    }
}

// ===========================================================================
// K4: in-chunk scan -> state; ssm = wC@state + D*xc; gate from w_gate@xp;
//     out = bn(w_out @ (sigmoid(gate)*(ssm+tanh(shift))))
// grid: NB*NCH, 256 threads (8 waves, one 16-px n-tile per wave)
// ===========================================================================
__global__ void k4_fused(const _Float16* __restrict__ xp,
                         const float* __restrict__ w_dw,
                         const float* __restrict__ g_conv,
                         const float* __restrict__ b_conv,
                         const float* __restrict__ wB,
                         const float* __restrict__ wC,
                         const float* __restrict__ A,
                         const float* __restrict__ Darr,
                         const float* __restrict__ w_gate,
                         const float* __restrict__ g_gate,
                         const float* __restrict__ b_gate,
                         const float* __restrict__ w_out,
                         const float* __restrict__ g_out,
                         const float* __restrict__ b_out,
                         const float* __restrict__ offsets,
                         float* __restrict__ out)
{
    __shared__ _Float16 xph[16][3][130];       // 12.2 KB
    __shared__ _Float16 xch[DINNER][CHUNK];    // 16 KB
    __shared__ float    state[DSTATE][CHUNK];  // 8 KB
    __shared__ _Float16 xg[DINNER][CHUNK];     // 16 KB
    __shared__ float    wBs[DSTATE * 64];      // 4 KB
    __shared__ float    wdws[64 * 9];
    __shared__ float    scc[64], bcc[64], asd[16];

    int tid = threadIdx.x;
    int bat = blockIdx.x >> 11;
    int ch  = blockIdx.x & (NCH - 1);
    int h   = ch >> 2;
    int w0  = (ch & (NCHROW - 1)) * CHUNK;
    float bnr = rsqrtf(1.f + 1e-5f);

    for (int idx = tid; idx < DSTATE * 64; idx += 256) wBs[idx] = wB[idx];
    for (int idx = tid; idx < 64 * 9; idx += 256) wdws[idx] = w_dw[idx];
    if (tid < 64) { scc[tid] = g_conv[tid] * bnr; bcc[tid] = b_conv[tid]; }
    if (tid < 16) asd[tid] = A[tid];
    __syncthreads();

    // --- recompute xc from xp (depthwise 3x3 + BN + SiLU), 16-ch groups ---
    for (int g = 0; g < 4; ++g) {
        for (int idx = tid; idx < 16 * 3 * 130; idx += 256) {
            int cg = idx / 390, r = idx % 390;
            int ry = r / 130, wx = r % 130;
            int c = g * 16 + cg;
            int hh = h + ry - 1, w = w0 + wx - 1;
            _Float16 v = (_Float16)0.f;
            if (hh >= 0 && hh < HH && w >= 0 && w < WWID)
                v = xp[(size_t)(bat * DINNER + c) * LL + (size_t)hh * WWID + w];
            xph[cg][ry][wx] = v;
        }
        __syncthreads();
        for (int idx = tid; idx < 16 * CHUNK; idx += 256) {
            int cg = idx >> 7, px = idx & 127;
            int c = g * 16 + cg;
            float s = 0.f;
#pragma unroll
            for (int dy = 0; dy < 3; ++dy)
#pragma unroll
                for (int dx = 0; dx < 3; ++dx)
                    s += (float)xph[cg][dy][px + dx] * wdws[c * 9 + dy * 3 + dx];
            s = silu(s * scc[c] + bcc[c]);
            xch[c][px] = (_Float16)s;
        }
        __syncthreads();
    }

    // --- weighted B projection per pixel ---
    if (tid < CHUNK) {
        int px = tid;
        float kf = (float)(h * WWID + w0 + px);
#pragma unroll
        for (int s = 0; s < DSTATE; ++s) {
            float acc = 0.f;
            for (int c = 0; c < 64; ++c)
                acc += wBs[s * 64 + c] * (float)xch[c][px];
            state[s][px] = acc * __expf(asd[s] * kf);
        }
    }
    __syncthreads();

    // --- in-chunk inclusive scan (Hillis-Steele over 128 px, 16 states) ---
    for (int off = 1; off < CHUNK; off <<= 1) {
        float t[DSTATE];
        if (tid < CHUNK) {
#pragma unroll
            for (int s = 0; s < DSTATE; ++s)
                t[s] = (tid >= off) ? state[s][tid - off] : 0.f;
        }
        __syncthreads();
        if (tid < CHUNK) {
#pragma unroll
            for (int s = 0; s < DSTATE; ++s)
                state[s][tid] += t[s];
        }
        __syncthreads();
    }
    if (tid < CHUNK) {
#pragma unroll
        for (int s = 0; s < DSTATE; ++s)
            state[s][tid] += offsets[(size_t)(bat * DSTATE + s) * NCH + ch];
    }
    __syncthreads();

    // --- WMMA phase: one 16-pixel n-tile per wave ---
    int l = tid & 31, wid = tid >> 5;
    int n = l & 15, half = (l >> 4) & 1;
    int px0 = wid * 16;
    int lane_px = px0 + n;

    // B fragment for state GEMM (K=16, zero-padded to 32)
    v16h bS;
#pragma unroll
    for (int e = 0; e < 16; ++e) {
        int K = kmapB(l, e);
        bS[e] = (K < DSTATE) ? (_Float16)state[K][lane_px] : (_Float16)0.f;
    }
    // B fragments for gate GEMM from xp center row (f16 in workspace, L2-hot)
    size_t rowbase = (size_t)bat * DINNER * LL + (size_t)h * WWID + w0 + lane_px;
    v16h bX0, bX1;
#pragma unroll
    for (int e = 0; e < 16; ++e) {
        int i0 = kmapB(l, e);
        bX0[e] = xp[rowbase + (size_t)i0 * LL];
        bX1[e] = xp[rowbase + (size_t)(32 + i0) * LL];
    }

#pragma unroll
    for (int t = 0; t < 4; ++t) {
        int row = t * 16 + n;
        v16h aC, aG0, aG1, aH0, aH1;
#pragma unroll
        for (int e = 0; e < 16; ++e) {
            int ka = kmapA(l, e);
            aC[e]  = (ka < DSTATE) ? (_Float16)wC[row * DSTATE + ka] : (_Float16)0.f;
            aG0[e] = (_Float16)w_gate[row * 64 + ka];
            aG1[e] = (_Float16)w_gate[row * 64 + 32 + ka];
            aH0[e] = (_Float16)w_gate[(64 + row) * 64 + ka];
            aH1[e] = (_Float16)w_gate[(64 + row) * 64 + 32 + ka];
        }
        v8f accS = {}, accG = {}, accH = {};
        accS = __builtin_amdgcn_wmma_f32_16x16x32_f16(false, aC,  false, bS,  (short)0, accS, false, false);
        accG = __builtin_amdgcn_wmma_f32_16x16x32_f16(false, aG0, false, bX0, (short)0, accG, false, false);
        accG = __builtin_amdgcn_wmma_f32_16x16x32_f16(false, aG1, false, bX1, (short)0, accG, false, false);
        accH = __builtin_amdgcn_wmma_f32_16x16x32_f16(false, aH0, false, bX0, (short)0, accH, false, false);
        accH = __builtin_amdgcn_wmma_f32_16x16x32_f16(false, aH1, false, bX1, (short)0, accH, false, false);
#pragma unroll
        for (int v = 0; v < 8; ++v) {
            int i  = t * 16 + half * 8 + v;
            float ssm  = accS[v] + Darr[i] * (float)xch[i][lane_px];
            float gate = accG[v] * (g_gate[i] * bnr) + b_gate[i];
            float shf  = accH[v] * (g_gate[64 + i] * bnr) + b_gate[64 + i];
            float xgv  = sigm(gate) * (ssm + tanhf(shf));
            xg[i][lane_px] = (_Float16)xgv;
        }
    }
    __syncthreads();

    // --- out projection 64x64 + BN ---
#pragma unroll
    for (int mt = 0; mt < 4; ++mt) {
        int row = mt * 16 + n;
        v16h a0, a1, bg0, bg1;
#pragma unroll
        for (int e = 0; e < 16; ++e) {
            int ka = kmapA(l, e);
            a0[e] = (_Float16)w_out[row * 64 + ka];
            a1[e] = (_Float16)w_out[row * 64 + 32 + ka];
            int kb = kmapB(l, e);
            bg0[e] = xg[kb][lane_px];
            bg1[e] = xg[32 + kb][lane_px];
        }
        v8f acc = {};
        acc = __builtin_amdgcn_wmma_f32_16x16x32_f16(false, a0, false, bg0, (short)0, acc, false, false);
        acc = __builtin_amdgcn_wmma_f32_16x16x32_f16(false, a1, false, bg1, (short)0, acc, false, false);
#pragma unroll
        for (int v = 0; v < 8; ++v) {
            int o = mt * 16 + half * 8 + v;
            float val = acc[v] * (g_out[o] * bnr) + b_out[o];
            out[(size_t)(bat * DMODEL + o) * LL + (size_t)h * WWID + w0 + lane_px] = val;
        }
    }
}

// ===========================================================================
extern "C" void kernel_launch(void* const* d_in, const int* in_sizes, int n_in,
                              void* d_out, int out_size, void* d_ws, size_t ws_size,
                              hipStream_t stream) {
    const float* x      = (const float*)d_in[0];
    const float* w_in   = (const float*)d_in[1];
    const float* g_in   = (const float*)d_in[2];
    const float* b_in   = (const float*)d_in[3];
    const float* w_dw   = (const float*)d_in[4];
    const float* g_conv = (const float*)d_in[5];
    const float* b_conv = (const float*)d_in[6];
    const float* wB     = (const float*)d_in[7];
    const float* wC     = (const float*)d_in[8];
    const float* A      = (const float*)d_in[9];
    const float* Darr   = (const float*)d_in[10];
    const float* w_gate = (const float*)d_in[11];
    const float* g_gate = (const float*)d_in[12];
    const float* b_gate = (const float*)d_in[13];
    const float* w_out  = (const float*)d_in[14];
    const float* g_out  = (const float*)d_in[15];
    const float* b_out  = (const float*)d_in[16];

    char* ws = (char*)d_ws;
    _Float16* xp     = (_Float16*)ws;                       // 268,435,456 B
    size_t xp_bytes  = (size_t)NB * DINNER * LL * sizeof(_Float16);
    float* partials  = (float*)(ws + xp_bytes);             // 1 MB
    float* offsets   = partials + (size_t)NB * DSTATE * NCH;

    dim3 blk(256);
    dim3 grid(NB * NCH);        // 16384
    k1_inproj<<<grid, blk, 0, stream>>>(x, w_in, g_in, b_in, xp);
    k2_partials<<<grid, blk, 0, stream>>>(xp, w_dw, g_conv, b_conv, wB, A, partials);
    k3_chunkscan<<<dim3(NB * DSTATE), blk, 0, stream>>>(partials, offsets);
    k4_fused<<<grid, blk, 0, stream>>>(xp, w_dw, g_conv, b_conv, wB, wC, A, Darr,
                                       w_gate, g_gate, b_gate, w_out, g_out, b_out,
                                       offsets, (float*)d_out);
}
